// SummarizeGCN_60258391163615
// MI455X (gfx1250) — matrix operand order repaired and checked by reading, not compiled, not run on hardware
//
#include <hip/hip_runtime.h>
#include <math.h>

#define NNODES 50000
#define NEDGES 800000
#define NC     128
#define NLAYER 4
#define NCLASS 40

typedef __attribute__((ext_vector_type(16))) __bf16 v16bf;
typedef __attribute__((ext_vector_type(8)))  __bf16 v8bf;
typedef __attribute__((ext_vector_type(8)))  float  v8f;

// ---------------- elementwise / prep kernels ----------------

__global__ void k_cvt_bf16(const float* __restrict__ in, __bf16* __restrict__ out, int n) {
  int t = blockIdx.x * blockDim.x + threadIdx.x;
  if (t < n) out[t] = (__bf16)in[t];
}

__global__ void k_fill1(float* __restrict__ p, int n) {
  int t = blockIdx.x * blockDim.x + threadIdx.x;
  if (t < n) p[t] = 1.0f;
}

__global__ void k_deg_edges(const long long* __restrict__ dst, float* __restrict__ deg, int ne) {
  int t = blockIdx.x * blockDim.x + threadIdx.x;
  if (t < ne) unsafeAtomicAdd(&deg[(int)dst[t]], 1.0f);
}

__global__ void k_rsqrt_inplace(float* __restrict__ p, int n) {
  int t = blockIdx.x * blockDim.x + threadIdx.x;
  if (t < n) p[t] = rsqrtf(p[t]);
}

// Pack a row-major 128x128 f32 weight into WMMA B-fragment lane order (bf16).
// Fragment (kt,nt): lane<16 -> col nt*16+lane, K kt*32+{0..7,16..23} ; lane>=16 -> K kt*32+{8..15,24..31}
__global__ void k_pack_b(const float* __restrict__ W, __bf16* __restrict__ P) {
  int p = blockIdx.x * blockDim.x + threadIdx.x;      // 0..16383
  if (p >= NC * NC) return;
  int i    = p & 15;
  int lane = (p >> 4) & 31;
  int nt   = (p >> 9) & 7;
  int kt   = p >> 12;
  int col  = nt * 16 + (lane & 15);
  int k    = kt * 32 + (lane >> 4) * 8 + (i & 7) + (i >> 3) * 16;
  P[p] = (__bf16)W[k * NC + col];
}

// ---------------- WMMA GEMM: C[rows x 128] = A[rows x 128] * B[128 x 128] ----------------
// One wave per 16-row tile; 8 column tiles; K loop 4 x 32.

__device__ __forceinline__ v16bf load_a_frag(const __bf16* __restrict__ arow, int kbase, int half) {
  v8bf lo = *(const v8bf*)(arow + kbase + half * 8);
  v8bf hi = *(const v8bf*)(arow + kbase + 16 + half * 8);
  return __builtin_shufflevector(lo, hi, 0,1,2,3,4,5,6,7,8,9,10,11,12,13,14,15);
}

__global__ void k_gemm_bf16(const __bf16* __restrict__ A, int lda,
                            const __bf16* __restrict__ Bp,
                            float* __restrict__ Cout, int ntiles) {
  int wave = blockIdx.x * (blockDim.x >> 5) + (threadIdx.x >> 5);
  if (wave >= ntiles) return;                    // wave-uniform exit: EXEC stays all-1 in live waves
  int lane = threadIdx.x & 31;
  int half = lane >> 4;
  int m    = lane & 15;
  const __bf16* arow = A + (size_t)(wave * 16 + m) * (size_t)lda;

  v8f acc[8];
#pragma unroll
  for (int i = 0; i < 8; ++i)
#pragma unroll
    for (int j = 0; j < 8; ++j) acc[i][j] = 0.0f;

#pragma unroll
  for (int kt = 0; kt < 4; ++kt) {
    v16bf afrag = load_a_frag(arow, kt * 32, half);
    const __bf16* btile = Bp + ((size_t)(kt * 8) * 32 + lane) * 16;
#pragma unroll
    for (int nt = 0; nt < 8; ++nt) {
      v16bf bfrag = *(const v16bf*)(btile + (size_t)nt * 32 * 16);
      acc[nt] = __builtin_amdgcn_wmma_f32_16x16x32_bf16(
          false, afrag, false, bfrag, (short)0, acc[nt], false, false);
    }
  }

  int rowbase = wave * 16 + half * 8;
#pragma unroll
  for (int nt = 0; nt < 8; ++nt)
#pragma unroll
    for (int r = 0; r < 8; ++r)
      Cout[(size_t)(rowbase + r) * NC + nt * 16 + m] = acc[nt][r];
}

// Attention score GEMM with fused epilogue: scores[row] = sum_n tanh((A*Watt)[row][n]) * a_att[n]
__global__ void k_att_score(const __bf16* __restrict__ A,
                            const __bf16* __restrict__ Bp,
                            const float* __restrict__ a_att,
                            float* __restrict__ scores, int ntiles) {
  int wave = blockIdx.x * (blockDim.x >> 5) + (threadIdx.x >> 5);
  if (wave >= ntiles) return;
  int lane = threadIdx.x & 31;
  int half = lane >> 4;
  int m    = lane & 15;
  const __bf16* arow = A + (size_t)(wave * 16 + m) * (size_t)NC;

  v8f acc[8];
#pragma unroll
  for (int i = 0; i < 8; ++i)
#pragma unroll
    for (int j = 0; j < 8; ++j) acc[i][j] = 0.0f;

#pragma unroll
  for (int kt = 0; kt < 4; ++kt) {
    v16bf afrag = load_a_frag(arow, kt * 32, half);
    const __bf16* btile = Bp + ((size_t)(kt * 8) * 32 + lane) * 16;
#pragma unroll
    for (int nt = 0; nt < 8; ++nt) {
      v16bf bfrag = *(const v16bf*)(btile + (size_t)nt * 32 * 16);
      acc[nt] = __builtin_amdgcn_wmma_f32_16x16x32_bf16(
          false, afrag, false, bfrag, (short)0, acc[nt], false, false);
    }
  }

  float sc[8];
#pragma unroll
  for (int r = 0; r < 8; ++r) sc[r] = 0.0f;
#pragma unroll
  for (int nt = 0; nt < 8; ++nt) {
    float av = a_att[nt * 16 + m];
#pragma unroll
    for (int r = 0; r < 8; ++r) sc[r] += tanhf(acc[nt][r]) * av;
  }
  // reduce over the 16 lanes of each half-wave (xor masks 1,2,4,8 stay within a 16-group)
#pragma unroll
  for (int off = 1; off < 16; off <<= 1)
#pragma unroll
    for (int r = 0; r < 8; ++r) sc[r] += __shfl_xor(sc[r], off, 32);

  if (m == 0) {
    int rowbase = wave * 16 + half * 8;
#pragma unroll
    for (int r = 0; r < 8; ++r) scores[rowbase + r] = sc[r];
  }
}

// ---------------- graph aggregation ----------------

__global__ void k_self_init(const float* __restrict__ hW, const float* __restrict__ dinv,
                            float* __restrict__ agg, int n) {
  int t = blockIdx.x * blockDim.x + threadIdx.x;
  if (t >= n) return;
  float di = dinv[t >> 7];
  agg[t] = hW[t] * di * di;
}

// one wave per edge, float4 per lane, hardware fp32 atomics
__global__ void k_edge_scatter(const float* __restrict__ hW, const float* __restrict__ dinv,
                               const long long* __restrict__ src, const long long* __restrict__ dst,
                               float* __restrict__ agg, int ne) {
  int t = blockIdx.x * blockDim.x + threadIdx.x;
  int e = t >> 5;
  if (e >= ne) return;
  int lane = t & 31;
  int s = (int)src[e];
  int d = (int)dst[e];
  float en = dinv[s] * dinv[d];
  const float4 v = *(const float4*)(hW + (size_t)s * NC + lane * 4);
  float* o = agg + (size_t)d * NC + lane * 4;
  unsafeAtomicAdd(o + 0, v.x * en);
  unsafeAtomicAdd(o + 1, v.y * en);
  unsafeAtomicAdd(o + 2, v.z * en);
  unsafeAtomicAdd(o + 3, v.w * en);
}

__global__ void k_relu_store(const float* __restrict__ agg, const float* __restrict__ bias,
                             __bf16* __restrict__ hbl, int n) {
  int t = blockIdx.x * blockDim.x + threadIdx.x;
  if (t >= n) return;
  int node = t >> 7, c = t & 127;
  float v = agg[t] + bias[c];
  v = v > 0.0f ? v : 0.0f;
  hbl[(size_t)node * (NLAYER * NC) + c] = (__bf16)v;   // [N, L, C] with layer offset baked into hbl
}

// ---------------- attention fuse + output head ----------------

__global__ void k_fuse(const float* __restrict__ scores, const __bf16* __restrict__ hb,
                       float* __restrict__ fused, int n) {
  int t = blockIdx.x * blockDim.x + threadIdx.x;
  if (t >= n) return;
  int node = t >> 7, c = t & 127;
  const float* s = scores + node * NLAYER;
  float s0 = s[0], s1 = s[1], s2 = s[2], s3 = s[3];
  float mx = fmaxf(fmaxf(s0, s1), fmaxf(s2, s3));
  float e0 = expf(s0 - mx), e1 = expf(s1 - mx), e2 = expf(s2 - mx), e3 = expf(s3 - mx);
  float inv = 1.0f / (e0 + e1 + e2 + e3);
  const __bf16* h = hb + (size_t)node * (NLAYER * NC) + c;
  fused[t] = inv * (e0 * (float)h[0] + e1 * (float)h[NC] + e2 * (float)h[2 * NC] + e3 * (float)h[3 * NC]);
}

__global__ void k_out(const float* __restrict__ fused, const float* __restrict__ Wout,
                      const float* __restrict__ bout, float* __restrict__ out, int n) {
  int t = blockIdx.x * blockDim.x + threadIdx.x;
  if (t >= n) return;
  int node = t / NCLASS;
  int c = t - node * NCLASS;
  const float* f = fused + (size_t)node * NC;
  float s = bout[c];
#pragma unroll 8
  for (int k = 0; k < NC; ++k) s = fmaf(f[k], Wout[k * NCLASS + c], s);
  out[t] = s;
}

// ---------------- host launcher ----------------

extern "C" void kernel_launch(void* const* d_in, const int* in_sizes, int n_in,
                              void* d_out, int out_size, void* d_ws, size_t ws_size,
                              hipStream_t stream) {
  const float*     x     = (const float*)d_in[0];
  const long long* ei    = (const long long*)d_in[1];
  const float*     convW = (const float*)d_in[2];
  const float*     convb = (const float*)d_in[3];
  const float*     W_att = (const float*)d_in[4];
  const float*     a_att = (const float*)d_in[5];
  const float*     W_out = (const float*)d_in[6];
  const float*     b_out = (const float*)d_in[7];
  const long long* src = ei;
  const long long* dst = ei + NEDGES;

  char* ws = (char*)d_ws;
  size_t off = 0;
  auto take = [&](size_t bytes) -> void* {
    void* p = ws + off;
    off = (off + bytes + 255) & ~(size_t)255;
    return p;
  };
  float*  dinv  = (float*) take((size_t)NNODES * 4);
  __bf16* xb    = (__bf16*)take((size_t)NNODES * NC * 2);
  __bf16* hb    = (__bf16*)take((size_t)NNODES * NLAYER * NC * 2);
  float*  hW    = (float*) take((size_t)NNODES * NC * 4);
  float*  agg   = (float*) take((size_t)NNODES * NC * 4);
  __bf16* pconv = (__bf16*)take((size_t)NLAYER * NC * NC * 2);
  __bf16* patt  = (__bf16*)take((size_t)NC * NC * 2);
  float*  fused  = hW;    // reuse: hW dead after last layer
  float*  scores = agg;   // reuse: agg dead after last relu (N*L*4 = 800KB << 25.6MB)

  const int TB = 256;
  const int nElem = NNODES * NC;

  // weight packing (tiny)
  for (int l = 0; l < NLAYER; ++l)
    k_pack_b<<<(NC * NC + TB - 1) / TB, TB, 0, stream>>>(convW + l * NC * NC, pconv + l * NC * NC);
  k_pack_b<<<(NC * NC + TB - 1) / TB, TB, 0, stream>>>(W_att, patt);

  // activations to bf16
  k_cvt_bf16<<<(nElem + TB - 1) / TB, TB, 0, stream>>>(x, xb, nElem);

  // degree -> dinv (in place)
  k_fill1<<<(NNODES + TB - 1) / TB, TB, 0, stream>>>(dinv, NNODES);
  k_deg_edges<<<(NEDGES + TB - 1) / TB, TB, 0, stream>>>(dst, dinv, NEDGES);
  k_rsqrt_inplace<<<(NNODES + TB - 1) / TB, TB, 0, stream>>>(dinv, NNODES);

  // GCN layers
  const int convTiles = NNODES / 16;               // 3125, exact
  const int gemmBlocks = (convTiles + 3) / 4;      // 4 waves (128 thr) per block
  const __bf16* Acur = xb;
  int lda = NC;
  for (int l = 0; l < NLAYER; ++l) {
    k_gemm_bf16<<<gemmBlocks, 128, 0, stream>>>(Acur, lda, pconv + l * NC * NC, hW, convTiles);
    k_self_init<<<(nElem + TB - 1) / TB, TB, 0, stream>>>(hW, dinv, agg, nElem);
    k_edge_scatter<<<(NEDGES * 32) / TB, TB, 0, stream>>>(hW, dinv, src, dst, agg, NEDGES);
    k_relu_store<<<(nElem + TB - 1) / TB, TB, 0, stream>>>(agg, convb + l * NC, hb + l * NC, nElem);
    Acur = hb + l * NC;
    lda = NLAYER * NC;
  }

  // layer attention + fuse + output head
  const int attTiles = (NNODES * NLAYER) / 16;     // 12500, exact
  k_att_score<<<(attTiles + 3) / 4, 128, 0, stream>>>(hb, patt, a_att, scores, attTiles);
  k_fuse<<<(nElem + TB - 1) / TB, TB, 0, stream>>>(scores, hb, fused, nElem);
  k_out<<<(NNODES * NCLASS + TB - 1) / TB, TB, 0, stream>>>(fused, W_out, b_out, (float*)d_out, NNODES * NCLASS);
}
